// Attention_86981677679328
// MI455X (gfx1250) — matrix operand order repaired and checked by reading, not compiled, not run on hardware
//
#include <hip/hip_runtime.h>
#include <hip/hip_bf16.h>

// ---------------------------------------------------------------------------
// Types for gfx1250 WMMA (wave32): A/B = 16 bf16 per lane, C/D = 8 f32 per lane
// ---------------------------------------------------------------------------
typedef __attribute__((ext_vector_type(16))) __bf16 v16bf;
typedef __attribute__((ext_vector_type(8)))  __bf16 v8bf;
typedef __attribute__((ext_vector_type(8)))  float  v8f;
typedef __attribute__((ext_vector_type(4)))  float  v4f;
typedef __attribute__((ext_vector_type(4)))  unsigned int u32x4;
typedef __attribute__((ext_vector_type(8)))  int  i32x8;
typedef __attribute__((ext_vector_type(4)))  int  i32x4;

__device__ __forceinline__ v8f wmma_bf16(v16bf a, v16bf b, v8f c) {
    return __builtin_amdgcn_wmma_f32_16x16x32_bf16(
        /*neg_a=*/false, a, /*neg_b=*/false, b,
        /*c_mod=*/(short)0, c, /*reuse_a=*/false, /*reuse_b=*/false);
}

__device__ __forceinline__ v16bf cat8(v8bf lo, v8bf hi) {
    return __builtin_shufflevector(lo, hi, 0,1,2,3,4,5,6,7,8,9,10,11,12,13,14,15);
}

// A-fragment (16x32, split-K run layout per ISA 7.12.2)
__device__ __forceinline__ v16bf frag_a_bf16(const __bf16* base, int row0, int k0,
                                             int ld, int lane) {
    int r   = row0 + (lane & 15);
    int klo = (lane >> 4) * 8;
    const __bf16* p = base + (size_t)r * ld + k0 + klo;
    v8bf lo = *reinterpret_cast<const v8bf*>(p);        // K = klo + 0..7
    v8bf hi = *reinterpret_cast<const v8bf*>(p + 16);   // K = 16 + klo + 0..7
    return cat8(lo, hi);
}

// A-fragment sourced from f32 (LDS scores) with on-the-fly bf16 convert
__device__ __forceinline__ v16bf frag_a_f32(const float* base, int row0, int k0,
                                            int ld, int lane) {
    int r   = row0 + (lane & 15);
    int klo = (lane >> 4) * 8;
    const float* p = base + (size_t)r * ld + k0 + klo;
    v4f a0 = *reinterpret_cast<const v4f*>(p);
    v4f a1 = *reinterpret_cast<const v4f*>(p + 4);
    v4f b0 = *reinterpret_cast<const v4f*>(p + 16);
    v4f b1 = *reinterpret_cast<const v4f*>(p + 20);
    v16bf o;
#pragma unroll
    for (int i = 0; i < 4; ++i) {
        o[i]      = (__bf16)a0[i];
        o[4 + i]  = (__bf16)a1[i];
        o[8 + i]  = (__bf16)b0[i];
        o[12 + i] = (__bf16)b1[i];
    }
    return o;
}

// B-fragment (32x16, split-K run layout matching A); base is B^T storage [n][k]
__device__ __forceinline__ v16bf frag_b_bf16(const __bf16* base, int col0, int k0,
                                             int ld, int lane) {
    int n  = col0 + (lane & 15);
    int kb = k0 + ((lane >> 4) << 3);
    const __bf16* p = base + (size_t)n * ld + kb;
    v8bf lo = *reinterpret_cast<const v8bf*>(p);
    v8bf hi = *reinterpret_cast<const v8bf*>(p + 16);
    return cat8(lo, hi);
}

// ---------------------------------------------------------------------------
// TDM: DMA a 2D bf16 tile (rows x 32 elems, src pitch 768 elems) into LDS with
// 32B pad after each 64B row -> 96B LDS pitch (matches stride-48 fragments).
//   D# group0: count=1 | lds_addr | global_addr | type=2
//   D# group1: data_size=2B, pad_enable, pad_interval=16dw, pad_amount=8dw,
//              huge tensor dims (tiles always in-bounds), tile 32 x rows,
//              dim0 stride = 768
//   groups 2/3 zero (2-D tile, iterate disabled); 6-arg builtin on this
//   toolchain takes an extra (zero) i32x8 before cpol.
// ---------------------------------------------------------------------------
__device__ __forceinline__ void tdm_load_tile(void* lds_dst, const void* gsrc,
                                              int rows) {
    uint64_t ga = (uint64_t)(uintptr_t)gsrc;
    u32x4 g0;
    g0[0] = 1u;                                            // count=1 (valid)
    g0[1] = (uint32_t)(uintptr_t)lds_dst;                  // LDS byte address
    g0[2] = (uint32_t)ga;                                  // global_addr[31:0]
    g0[3] = ((uint32_t)(ga >> 32) & 0x01FFFFFFu) | (2u << 30); // addr[56:32]|type=2
    i32x8 g1;
    g1[0] = (1 << 16) | (1 << 20) | (3 << 22) | (7 << 25); // 2B, pad 16dw->+8dw
    g1[1] = 0;                                             // (tensor_dim0 lo16=0)
    g1[2] = 0x0010;                                        // tensor_dim0 = 1M
    g1[3] = 0x0010 | (32 << 16);                           // tensor_dim1=1M, tile_dim0=32
    g1[4] = rows;                                          // tile_dim1
    g1[5] = 768;                                           // tensor_dim0_stride
    g1[6] = 0;
    g1[7] = 0;
    i32x4 z4 = {0, 0, 0, 0};
    i32x8 z8 = {0, 0, 0, 0, 0, 0, 0, 0};
    __builtin_amdgcn_tensor_load_to_lds(g0, g1, z4, z4, z8, 0);
}

// ---------------------------------------------------------------------------
// Prep kernels: one-time f32 -> bf16 convert, and convert+transpose weights
// ---------------------------------------------------------------------------
__global__ __launch_bounds__(256) void cvt_bf16_kernel(const float* __restrict__ in,
                                                       __bf16* __restrict__ out) {
    int i = (blockIdx.x * 256 + threadIdx.x) * 4;
    v4f f = *reinterpret_cast<const v4f*>(in + i);
#pragma unroll
    for (int j = 0; j < 4; ++j) out[i + j] = (__bf16)f[j];
}

__global__ __launch_bounds__(256) void transpose_cvt_kernel(
    const float* __restrict__ in /*[K][N]*/, __bf16* __restrict__ out /*[N][K]*/,
    int Kd, int Nd) {
    __shared__ float tile[32][33];
    const int n0 = blockIdx.x * 32, k0 = blockIdx.y * 32;
    const int tx = threadIdx.x & 31, ty = threadIdx.x >> 5;  // 32 x 8
#pragma unroll
    for (int i = 0; i < 32; i += 8)
        tile[ty + i][tx] = in[(size_t)(k0 + ty + i) * Nd + n0 + tx];
    __syncthreads();
#pragma unroll
    for (int i = 0; i < 32; i += 8)
        out[(size_t)(n0 + ty + i) * Kd + k0 + tx] = (__bf16)tile[tx][ty + i];
}

// ---------------------------------------------------------------------------
// GEMM: C[M,Nc] = A[M,768]_bf16 * Wt[Nc,768]_bf16^T + bias (f32 accum)
// Block tile 128(M) x 64(N), 8 waves x 4 M-subtiles.  Tiles DMA'd by the TDM
// (wave 0 issues, double-buffered), compute waves never touch global memory.
//   mode 0: Nc=2304 -> scatter Q (pre-scaled) / K as [B,H,N,64], V as [B,H,64,N]
//   mode 1: Nc=768  -> f32 output + bias
// ---------------------------------------------------------------------------
__global__ __launch_bounds__(256) void gemm_bf16_kernel(
    const __bf16* __restrict__ A, const __bf16* __restrict__ Wt,
    const float* __restrict__ bias, int K, int Nc, int mode,
    __bf16* __restrict__ Qo, __bf16* __restrict__ Ko, __bf16* __restrict__ Vo,
    float* __restrict__ Out) {
    __shared__ __attribute__((aligned(16))) __bf16 At[2][128 * 48];
    __shared__ __attribute__((aligned(16))) __bf16 Bt[2][64 * 48];

    const int t = threadIdx.x;
    const int w = t >> 5, l = t & 31;
    const int n0 = blockIdx.x * 64;
    const int m0 = blockIdx.y * 128;
    const int nsub = (w & 3) * 16;
    const int mq   = (w >> 2) * 64;

    const __bf16* ag = A  + (size_t)m0 * K;   // tile origin, advances 32 elem/step
    const __bf16* bg = Wt + (size_t)n0 * K;
    const bool issuer = (w == 0);
    const int nsteps = K / 32;

    if (issuer) {
        tdm_load_tile(At[0], ag, 128);
        tdm_load_tile(Bt[0], bg, 64);
    }

    v8f acc[4] = {{}, {}, {}, {}};

    for (int s = 0; s < nsteps; ++s) {
        const int cur = s & 1;
        if (issuer) {
            if (s + 1 < nsteps) {
                tdm_load_tile(At[cur ^ 1], ag + (size_t)(s + 1) * 32, 128);
                tdm_load_tile(Bt[cur ^ 1], bg + (size_t)(s + 1) * 32, 64);
                __builtin_amdgcn_s_wait_tensorcnt(2);  // current pair complete
            } else {
                __builtin_amdgcn_s_wait_tensorcnt(0);
            }
        }
        __syncthreads();
        v16bf bf = frag_b_bf16(Bt[cur], nsub, 0, 48, l);
#pragma unroll
        for (int mi = 0; mi < 4; ++mi) {
            v16bf a = frag_a_bf16(At[cur], mq + mi * 16, 0, 48, l);
            acc[mi] = wmma_bf16(a, bf, acc[mi]);
        }
        __syncthreads();
    }

    // Epilogue. C/D layout: VGPR r -> M = r + (l>=16)*8, N = l&15.
    const int c  = n0 + nsub + (l & 15);
    const float bv = bias[c];
    const int rb = (l >> 4) * 8;
#pragma unroll
    for (int mi = 0; mi < 4; ++mi) {
#pragma unroll
        for (int r = 0; r < 8; ++r) {
            float v = acc[mi][r] + bv;
            int m = m0 + mq + mi * 16 + rb + r;
            if (mode == 0) {
                int which = c / 768, c2 = c % 768;
                int h = c2 >> 6, d = c2 & 63;
                int b = m >> 10, n = m & 1023;
                int bh = b * 12 + h;
                if (which == 0)
                    Qo[((size_t)bh * 1024 + n) * 64 + d] = (__bf16)(v * 0.125f);
                else if (which == 1)
                    Ko[((size_t)bh * 1024 + n) * 64 + d] = (__bf16)v;
                else  // V stored transposed: [bh][d][n]
                    Vo[((size_t)bh * 64 + d) * 1024 + n] = (__bf16)v;
            } else {
                Out[(size_t)m * Nc + c] = v;
            }
        }
    }
}

// ---------------------------------------------------------------------------
// Attention for one (b, h, 64-query chunk).  Full 64x1024 f32 score tile in
// LDS (~258 KB of the 320 KB WGP pool) -> exact 3-barrier softmax.  PV reads
// probabilities from LDS and V^T straight from global; output written bf16.
// ---------------------------------------------------------------------------
#define S_LD 1032   // f32 row stride (1032%64==8 dwords: 2-way conflicts max)

__global__ __launch_bounds__(256) void attn_kernel(
    const __bf16* __restrict__ Q, const __bf16* __restrict__ Kb,
    const __bf16* __restrict__ Vt, __bf16* __restrict__ O /* [B*N,768] bf16 */) {
    extern __shared__ char smem[];
    float* S    = reinterpret_cast<float*>(smem);                  // [64][1032]
    float* pmax = reinterpret_cast<float*>(smem + 64 * S_LD * 4);  // [64][4]
    float* psum = pmax + 256;                                      // [64][4]

    const int t = threadIdx.x;
    const int w = t >> 5, l = t & 31;
    const int bh = blockIdx.y;
    const int b = bh / 12, h = bh % 12;
    const int q0 = blockIdx.x * 64;

    const __bf16* qb  = Q  + ((size_t)bh * 1024 + q0) * 64;  // [64 q][64 d]
    const __bf16* kb  = Kb + (size_t)bh * 1024 * 64;         // [1024 k][64 d]
    const __bf16* vtb = Vt + (size_t)bh * 64 * 1024;         // [64 d][1024 k]

    // ---- Phase 1: S = (Q*scale) K^T ; wave w owns key columns [w*128, +128)
    v16bf aq[4][2];
#pragma unroll
    for (int mi = 0; mi < 4; ++mi)
#pragma unroll
        for (int d = 0; d < 2; ++d)
            aq[mi][d] = frag_a_bf16(qb, mi * 16, d * 32, 64, l);

    const int nw = w * 128;
#pragma unroll
    for (int nj = 0; nj < 8; ++nj) {
        int kr = nw + nj * 16;
        v16bf b0 = frag_b_bf16(kb, kr, 0, 64, l);
        v16bf b1 = frag_b_bf16(kb, kr, 32, 64, l);
#pragma unroll
        for (int mi = 0; mi < 4; ++mi) {
            v8f acc = {};
            acc = wmma_bf16(aq[mi][0], b0, acc);
            acc = wmma_bf16(aq[mi][1], b1, acc);
            int col   = kr + (l & 15);
            int rbase = mi * 16 + ((l >> 4) * 8);
#pragma unroll
            for (int r = 0; r < 8; ++r) S[(size_t)(rbase + r) * S_LD + col] = acc[r];
        }
    }
    __syncthreads();

    // ---- Phase 2: exact softmax; 4 threads per row, 256 cols each
    {
        int row = t >> 2, qtr = t & 3;
        float* sr = S + (size_t)row * S_LD + qtr * 256;
        float m = -1e30f;
        for (int i = 0; i < 256; ++i) m = fmaxf(m, sr[i]);
        pmax[row * 4 + qtr] = m;
        __syncthreads();
        float rm = fmaxf(fmaxf(pmax[row * 4 + 0], pmax[row * 4 + 1]),
                         fmaxf(pmax[row * 4 + 2], pmax[row * 4 + 3]));
        float s = 0.f;
        for (int i = 0; i < 256; ++i) { float e = __expf(sr[i] - rm); sr[i] = e; s += e; }
        psum[row * 4 + qtr] = s;
        __syncthreads();
        float inv = 1.f / (psum[row * 4 + 0] + psum[row * 4 + 1] +
                           psum[row * 4 + 2] + psum[row * 4 + 3]);
        for (int i = 0; i < 256; ++i) sr[i] *= inv;
    }
    __syncthreads();

    // ---- Phase 3: O = P @ V.  P from LDS (f32->bf16), V^T from global.
    v8f o0 = {}, o1 = {};
    const int nsub = (w & 3) * 16;
    const int mt0  = (w >> 2) * 32;
#pragma unroll 4
    for (int ks = 0; ks < 32; ++ks) {
        v16bf bf = frag_b_bf16(vtb, nsub, ks * 32, 1024, l);
        v16bf a0 = frag_a_f32(S, mt0,      ks * 32, S_LD, l);
        v16bf a1 = frag_a_f32(S, mt0 + 16, ks * 32, S_LD, l);
        o0 = wmma_bf16(a0, bf, o0);
        o1 = wmma_bf16(a1, bf, o1);
    }

    // ---- Epilogue: bf16 activation [B*N, 768] for the proj GEMM
    const int col = h * 64 + nsub + (l & 15);
    const int rb  = (l >> 4) * 8;
#pragma unroll
    for (int r = 0; r < 8; ++r) {
        int mg = q0 + mt0 + rb + r;
        O[((size_t)b * 1024 + mg) * 768 + col]      = (__bf16)o0[r];
        O[((size_t)b * 1024 + mg + 16) * 768 + col] = (__bf16)o1[r];
    }
}

// ---------------------------------------------------------------------------
// Launch
// ---------------------------------------------------------------------------
extern "C" void kernel_launch(void* const* d_in, const int* in_sizes, int n_in,
                              void* d_out, int out_size, void* d_ws, size_t ws_size,
                              hipStream_t stream) {
    const float* x      = (const float*)d_in[0];   // [8,1024,768]
    const float* w_qkv  = (const float*)d_in[1];   // [768,2304]
    const float* b_qkv  = (const float*)d_in[2];   // [2304]
    const float* w_proj = (const float*)d_in[3];   // [768,768]
    const float* b_proj = (const float*)d_in[4];   // [768]
    float* out = (float*)d_out;                    // [8,1024,768]

    char* ws = (char*)d_ws;
    const size_t x_b    = (size_t)8192 * 768 * 2;        // 12.58 MB
    const size_t wqkv_b = (size_t)2304 * 768 * 2;        //  3.54 MB
    const size_t wp_b   = (size_t)768 * 768 * 2;         //  1.18 MB
    __bf16* Xb    = (__bf16*)(ws);
    __bf16* Wqkvt = (__bf16*)(ws + x_b);
    __bf16* Wpt   = (__bf16*)(ws + x_b + wqkv_b);
    __bf16* Qw    = (__bf16*)(ws + x_b + wqkv_b + wp_b);
    __bf16* Kw    = (__bf16*)(ws + x_b + wqkv_b + wp_b + x_b);
    __bf16* Vw    = (__bf16*)(ws + x_b + wqkv_b + wp_b + 2 * x_b);  // [bh][d][n]
    __bf16* Awb   = (__bf16*)(ws + x_b + wqkv_b + wp_b + 3 * x_b);  // [8192,768]

    dim3 blk(256);

    // 0) one-time bf16 conversions / weight transposes
    cvt_bf16_kernel<<<dim3(8192 * 768 / (256 * 4)), blk, 0, stream>>>(x, Xb);
    transpose_cvt_kernel<<<dim3(2304 / 32, 768 / 32), blk, 0, stream>>>(
        w_qkv, Wqkvt, 768, 2304);
    transpose_cvt_kernel<<<dim3(768 / 32, 768 / 32), blk, 0, stream>>>(
        w_proj, Wpt, 768, 768);

    // 1) QKV projection (TDM-fed) -> bf16 Q (pre-scaled) / K / V^T per head
    gemm_bf16_kernel<<<dim3(2304 / 64, 8192 / 128), blk, 0, stream>>>(
        Xb, Wqkvt, b_qkv, 768, 2304, /*mode=*/0, Qw, Kw, Vw, nullptr);

    // 2) attention per (b,h,64-row query chunk); big-LDS softmax
    const size_t smem = (size_t)64 * S_LD * 4 + 512 * 4;
    attn_kernel<<<dim3(1024 / 64, 8 * 12), blk, smem, stream>>>(Qw, Kw, Vw, Awb);

    // 3) output projection (TDM-fed) -> d_out
    gemm_bf16_kernel<<<dim3(768 / 64, 8192 / 128), blk, 0, stream>>>(
        Awb, Wpt, b_proj, 768, 768, /*mode=*/1, nullptr, nullptr, nullptr, out);
}
